// CEN_BRL_52810917872041
// MI455X (gfx1250) — compile-verified, matrix operand order checked by reading, not run on hardware
//
#include <hip/hip_runtime.h>
#include <hip/hip_bf16.h>
#include <math.h>

typedef __attribute__((ext_vector_type(16))) _Float16 v16h;
typedef __attribute__((ext_vector_type(8)))  float    v8f;

#define N_TRAIN 4096
#define N_FEAT  64
#define N_HID   256
#define ENC     256
#define GH      256
#define AH      64
#define N_ANTES 32768
#define MAX_LEN 32

__device__ __forceinline__ float sigmoidf_(float x) { return 1.0f / (1.0f + __expf(-x)); }

// ---------------------------------------------------------------------------
// Fragment loaders for v_wmma_f32_16x16x32_f16 (ISA 7.12.2 layouts).
// A 16x32 (MxK): lane<16 -> M=lane, halves 0..7 = K koff+0..7, 8..15 = K koff+16..23
//                (koff = 0 for lanes 0..15, 8 for lanes 16..31)
// B 32x16 (KxN): loaded as B^T with the A pattern, lane = output column N.
// ---------------------------------------------------------------------------
__device__ __forceinline__ v16h cvt_frag(float4 x0, float4 x1, float4 y0, float4 y1)
{
    v16h a;
    a[0]=(_Float16)x0.x;  a[1]=(_Float16)x0.y;  a[2]=(_Float16)x0.z;  a[3]=(_Float16)x0.w;
    a[4]=(_Float16)x1.x;  a[5]=(_Float16)x1.y;  a[6]=(_Float16)x1.z;  a[7]=(_Float16)x1.w;
    a[8]=(_Float16)y0.x;  a[9]=(_Float16)y0.y;  a[10]=(_Float16)y0.z; a[11]=(_Float16)y0.w;
    a[12]=(_Float16)y1.x; a[13]=(_Float16)y1.y; a[14]=(_Float16)y1.z; a[15]=(_Float16)y1.w;
    return a;
}

// row-major per-lane: two runs of 8 contiguous floats -> 4x global_load_b128
__device__ __forceinline__ v16h load_frag_rows(const float* __restrict__ base)
{
    const float4* p0 = (const float4*)base;         // K = koff+0..7
    const float4* p1 = (const float4*)(base + 16);  // K = koff+16..23
    return cvt_frag(p0[0], p0[1], p1[0], p1[1]);
}

// K-major (strided) per-lane: lane-coalesced scalar loads
__device__ __forceinline__ v16h load_frag_cols(const float* __restrict__ B, int ldb,
                                               int col, int k0, int koff)
{
    v16h b;
#pragma unroll
    for (int p = 0; p < 16; ++p) {
        const int k = k0 + koff + ((p < 8) ? p : (8 + p)); // p>=8 -> 16+(p-8)
        b[p] = (_Float16)B[(size_t)k * ldb + col];
    }
    return b;
}

// ---------------------------------------------------------------------------
// C[M,N] = op(A)[M,K] * op(B)[K,N] (+bias)(+addm)(relu)
// One wave computes an (MT*16) x (NT*16) macro-tile; A fragments are reused
// across NT B tiles and vice versa -> MT*NT WMMAs per 32-wide K chunk.
// PF: prefetch the next A K-chunk (only worthwhile for the HBM-streaming GEMM).
// ---------------------------------------------------------------------------
template<bool ATRANS, bool BTRANS, bool HASBIAS, bool HASADD, bool RELU,
         int MT, int NT, bool PF>
__global__ void gemm_wmma(const float* __restrict__ A, int lda,
                          const float* __restrict__ B, int ldb,
                          const float* __restrict__ bias,
                          const float* __restrict__ addm, int ldadd,
                          float* __restrict__ C, int ldc,
                          int M, int N, int K)
{
    const int lane = threadIdx.x;                  // 0..31 (wave32)
    const int tilesN = N / (16 * NT);
    const int tile = blockIdx.x * blockDim.y + threadIdx.y;
    if (tile >= (M / (16 * MT)) * tilesN) return;
    const int tm = tile / tilesN;
    const int tn = tile - tm * tilesN;
    const int row0 = tm * 16 * MT, col0 = tn * 16 * NT;
    const int l15  = lane & 15;
    const int koff = (lane >> 4) << 3;             // 0 or 8

    v8f acc[MT][NT] = {};

    for (int k0 = 0; k0 < K; k0 += 32) {
        if (PF && (k0 + 32 < K)) {  // global_prefetch_b8 of the next A chunk
#pragma unroll
            for (int mt = 0; mt < MT; ++mt) {
                if (ATRANS) __builtin_prefetch(A + (size_t)(k0 + 32 + koff) * lda + row0 + 16 * mt + l15, 0, 3);
                else        __builtin_prefetch(A + (size_t)(row0 + 16 * mt + l15) * lda + k0 + 32 + koff, 0, 3);
            }
        }

        v16h a[MT], b[NT];
#pragma unroll
        for (int mt = 0; mt < MT; ++mt) {
            if (ATRANS) a[mt] = load_frag_cols(A, lda, row0 + 16 * mt + l15, k0, koff);
            else        a[mt] = load_frag_rows(A + (size_t)(row0 + 16 * mt + l15) * lda + k0 + koff);
        }
#pragma unroll
        for (int nt = 0; nt < NT; ++nt) {
            if (BTRANS) b[nt] = load_frag_rows(B + (size_t)(col0 + 16 * nt + l15) * ldb + k0 + koff);
            else        b[nt] = load_frag_cols(B, ldb, col0 + 16 * nt + l15, k0, koff);
        }

#pragma unroll
        for (int mt = 0; mt < MT; ++mt)
#pragma unroll
            for (int nt = 0; nt < NT; ++nt)
                acc[mt][nt] = __builtin_amdgcn_wmma_f32_16x16x32_f16(
                                  false, a[mt], false, b[nt], (short)0,
                                  acc[mt][nt], false, false);
    }

    const int mb = (lane >> 4) << 3;               // C/D: VGPR r -> M = mb + r
#pragma unroll
    for (int mt = 0; mt < MT; ++mt) {
#pragma unroll
        for (int nt = 0; nt < NT; ++nt) {
            const int col = col0 + 16 * nt + l15;
#pragma unroll
            for (int r = 0; r < 8; ++r) {
                const int row = row0 + 16 * mt + mb + r;
                float v = acc[mt][nt][r];
                if (HASBIAS) v += bias[col];
                if (HASADD)  v += addm[(size_t)row * ldadd + col];
                if (RELU)    v = fmaxf(v, 0.0f);
                C[(size_t)row * ldc + col] = v;
            }
        }
    }
}

// ---------------------------------------------------------------------------
__global__ void zero_f32(float* __restrict__ p, int n)
{
    int i = blockIdx.x * blockDim.x + threadIdx.x;
    if (i < n) p[i] = 0.0f;
}

__global__ void add_vec(const float* __restrict__ a, const float* __restrict__ b,
                        float* __restrict__ o, int n)
{
    int i = blockIdx.x * blockDim.x + threadIdx.x;
    if (i < n) o[i] = a[i] + b[i];
}

// LSTM cell elementwise: gates[m, 0:4*GH] -> (h, c) update
__global__ void lstm_cell(const float* __restrict__ gates,
                          float* __restrict__ h, float* __restrict__ c)
{
    const int idx = blockIdx.x * blockDim.x + threadIdx.x; // < N_TRAIN*GH
    const int m = idx >> 8;      // / GH (GH==256)
    const int n = idx & (GH - 1);
    const size_t gb = (size_t)m * (4 * GH);
    const float ig = sigmoidf_(gates[gb + n]);
    const float fg = sigmoidf_(gates[gb + GH + n]);
    const float gg = tanhf(gates[gb + 2 * GH + n]);
    const float og = sigmoidf_(gates[gb + 3 * GH + n]);
    const float cn = fg * c[idx] + ig * gg;
    c[idx] = cn;
    h[idx] = og * tanhf(cn);
}

// e_t[n] = mean over m of h[m, n]  (one block per column)
__global__ void col_mean(const float* __restrict__ h, float* __restrict__ e_t)
{
    __shared__ float sm[256];
    const int n = blockIdx.x;
    float s = 0.0f;
    for (int m = threadIdx.x; m < N_TRAIN; m += blockDim.x)
        s += h[(size_t)m * GH + n];
    sm[threadIdx.x] = s;
    __syncthreads();
    for (int off = blockDim.x >> 1; off > 0; off >>= 1) {
        if ((int)threadIdx.x < off) sm[threadIdx.x] += sm[threadIdx.x + off];
        __syncthreads();
    }
    if (threadIdx.x == 0) e_t[n] = sm[0] * (1.0f / (float)N_TRAIN);
}

// avec[j] = sum_n e_t[n] * W1_h[n, j] + att_b1[j]   (W1_h = att_W1[N_TRAIN:])
__global__ void att_vec(const float* __restrict__ e_t,
                        const float* __restrict__ att_W1,
                        const float* __restrict__ att_b1,
                        float* __restrict__ avec)
{
    const int j = threadIdx.x;   // 0..AH-1
    const float* W1h = att_W1 + (size_t)N_TRAIN * AH;
    float s = att_b1[j];
    for (int n = 0; n < GH; ++n) s += e_t[n] * W1h[(size_t)n * AH + j];
    avec[j] = s;
}

// scores[a] = relu(S_proj[a,:] + avec) . att_w2 + att_b2[0]  (one wave per ante)
__global__ void att_scores(const float* __restrict__ Sproj,
                           const float* __restrict__ avec,
                           const float* __restrict__ att_w2,
                           const float* __restrict__ att_b2,
                           float* __restrict__ scores_t)
{
    const int lane = threadIdx.x;                       // 0..31
    const int a = blockIdx.x * blockDim.y + threadIdx.y;
    const float2 sp = ((const float2*)(Sproj + (size_t)a * AH))[lane];
    const float2 av = ((const float2*)avec)[lane];
    const float2 w2 = ((const float2*)att_w2)[lane];
    float v = fmaxf(sp.x + av.x, 0.0f) * w2.x + fmaxf(sp.y + av.y, 0.0f) * w2.y;
    for (int off = 16; off > 0; off >>= 1) v += __shfl_down(v, off, 32);
    if (lane == 0) scores_t[a] = v + att_b2[0];
}

// per-step argmax over N_ANTES (one block per step), first-occurrence ties
__global__ void argmax_rows(const float* __restrict__ scores, int* __restrict__ idx_out)
{
    __shared__ float sv[256];
    __shared__ int   si[256];
    const int t = blockIdx.x;
    const float* row = scores + (size_t)t * N_ANTES;
    float best = -INFINITY; int bi = 0x7fffffff;
    for (int a = threadIdx.x; a < N_ANTES; a += blockDim.x) {
        float v = row[a];
        if (v > best || (v == best && a < bi)) { best = v; bi = a; }
    }
    sv[threadIdx.x] = best; si[threadIdx.x] = bi;
    __syncthreads();
    for (int off = 128; off > 0; off >>= 1) {
        if ((int)threadIdx.x < off) {
            if (sv[threadIdx.x + off] > sv[threadIdx.x] ||
                (sv[threadIdx.x + off] == sv[threadIdx.x] &&
                 si[threadIdx.x + off] < si[threadIdx.x])) {
                sv[threadIdx.x] = sv[threadIdx.x + off];
                si[threadIdx.x] = si[threadIdx.x + off];
            }
        }
        __syncthreads();
    }
    if (threadIdx.x == 0) idx_out[t] = si[0];
}

// ---------------------------------------------------------------------------
static inline int gemmGrid(int M, int N, int MT, int NT)
{
    return ((M / (16 * MT)) * (N / (16 * NT)) + 3) / 4;
}

extern "C" void kernel_launch(void* const* d_in, const int* in_sizes, int n_in,
                              void* d_out, int out_size, void* d_ws, size_t ws_size,
                              hipStream_t stream)
{
    const float* context = (const float*)d_in[0];   // (4096, 64)
    const float* S       = (const float*)d_in[1];   // (4096, 32768)
    const float* enc_W1  = (const float*)d_in[2];   // (64, 256)
    const float* enc_b1  = (const float*)d_in[3];   // (256,)
    const float* enc_W2  = (const float*)d_in[4];   // (256, 256)
    const float* enc_b2  = (const float*)d_in[5];   // (256,)
    const float* W_ih    = (const float*)d_in[6];   // (1024, 256)
    const float* W_hh    = (const float*)d_in[7];   // (1024, 256)
    const float* b_ih    = (const float*)d_in[8];   // (1024,)
    const float* b_hh    = (const float*)d_in[9];   // (1024,)
    const float* att_W1  = (const float*)d_in[10];  // (4096+256, 64)
    const float* att_b1  = (const float*)d_in[11];  // (64,)
    const float* att_w2  = (const float*)d_in[12];  // (64,)
    const float* att_b2  = (const float*)d_in[13];  // (1,)
    // d_in[14] = max_len scalar (device); compile-time MAX_LEN used instead.

    float* ws = (float*)d_ws;
    // workspace layout (floats)
    float* xproj = ws;                                   // 4096*1024
    float* Sproj = xproj + (size_t)N_TRAIN * 4 * GH;     // 32768*64
    float* h     = Sproj + (size_t)N_ANTES * AH;         // 4096*256
    float* c     = h     + (size_t)N_TRAIN * GH;         // 4096*256
    float* e_t   = c     + (size_t)N_TRAIN * GH;         // 256
    float* avec  = e_t + GH;                             // 64
    float* biasc = avec + AH;                            // 1024
    float* tmp   = biasc + 4 * GH;                       // 4096*1024 (h1/phi, then gates)
    float* h1    = tmp;                                  // 4096*256
    float* phi   = tmp + (size_t)N_TRAIN * ENC;          // 4096*256
    float* gates = tmp;                                  // 4096*1024 (reuses h1/phi)

    float* scores  = (float*)d_out;                      // 32 * 32768
    int*   idx_out = (int*)d_out + (size_t)MAX_LEN * N_ANTES;

    dim3 gblock(32, 4);   // 4 waves / block, one macro-tile per wave

    // combined LSTM input bias
    add_vec<<<(4 * GH + 255) / 256, 256, 0, stream>>>(b_ih, b_hh, biasc, 4 * GH);

    // h1 = relu(context @ enc_W1 + enc_b1)          [4096 x 256, K=64]
    gemm_wmma<false, false, true, false, true, 2, 2, false>
        <<<gemmGrid(N_TRAIN, N_HID, 2, 2), gblock, 0, stream>>>(
        context, N_FEAT, enc_W1, N_HID, enc_b1, nullptr, 0,
        h1, N_HID, N_TRAIN, N_HID, N_FEAT);

    // phi = h1 @ enc_W2 + enc_b2                    [4096 x 256, K=256]
    gemm_wmma<false, false, true, false, false, 2, 2, false>
        <<<gemmGrid(N_TRAIN, ENC, 2, 2), gblock, 0, stream>>>(
        h1, N_HID, enc_W2, ENC, enc_b2, nullptr, 0,
        phi, ENC, N_TRAIN, ENC, N_HID);

    // x_proj = phi @ W_ih^T + (b_ih + b_hh)         [4096 x 1024, K=256]
    gemm_wmma<false, true, true, false, false, 2, 2, false>
        <<<gemmGrid(N_TRAIN, 4 * GH, 2, 2), gblock, 0, stream>>>(
        phi, ENC, W_ih, ENC, biasc, nullptr, 0,
        xproj, 4 * GH, N_TRAIN, 4 * GH, ENC);

    // S_proj = S^T @ att_W1[:N_TRAIN]               [32768 x 64, K=4096]
    // NT=4 -> one wave covers all 64 columns -> S (512 MB) is read exactly once.
    // MT=1 keeps VGPR pressure low (no accumulator spills); MT does not change
    // HBM traffic here since tilesN==1. PF -> global_prefetch_b8 on the S stream.
    gemm_wmma<true, false, false, false, false, 1, 4, true>
        <<<gemmGrid(N_ANTES, AH, 1, 4), gblock, 0, stream>>>(
        S, N_ANTES, att_W1, AH, nullptr, nullptr, 0,
        Sproj, AH, N_ANTES, AH, N_TRAIN);

    // h0 = c0 = 0
    zero_f32<<<(N_TRAIN * GH + 255) / 256, 256, 0, stream>>>(h, N_TRAIN * GH);
    zero_f32<<<(N_TRAIN * GH + 255) / 256, 256, 0, stream>>>(c, N_TRAIN * GH);

    for (int t = 0; t < MAX_LEN; ++t) {
        // gates = x_proj + h @ W_hh^T               [4096 x 1024, K=256]
        gemm_wmma<false, true, false, true, false, 2, 2, false>
            <<<gemmGrid(N_TRAIN, 4 * GH, 2, 2), gblock, 0, stream>>>(
            h, GH, W_hh, GH, nullptr, xproj, 4 * GH,
            gates, 4 * GH, N_TRAIN, 4 * GH, GH);

        lstm_cell<<<(N_TRAIN * GH) / 256, 256, 0, stream>>>(gates, h, c);

        col_mean<<<GH, 256, 0, stream>>>(h, e_t);

        att_vec<<<1, AH, 0, stream>>>(e_t, att_W1, att_b1, avec);

        att_scores<<<N_ANTES / 8, dim3(32, 8), 0, stream>>>(
            Sproj, avec, att_w2, att_b2, scores + (size_t)t * N_ANTES);
    }

    argmax_rows<<<MAX_LEN, 256, 0, stream>>>(scores, idx_out);
}
